// MultiTokenPredictionHead_4363686773541
// MI455X (gfx1250) — matrix operand order repaired and checked by reading, not compiled, not run on hardware
//
#include <hip/hip_runtime.h>
#include <stdint.h>

// ---------------------------------------------------------------------------
// MI455X / gfx1250 (CDNA5, wave32) implementation.
// Matrix math: v_wmma_f32_16x16x32_bf16. Tile staging: async global->LDS.
// ---------------------------------------------------------------------------

typedef __attribute__((ext_vector_type(16))) __bf16 v16bf;
typedef __attribute__((ext_vector_type(8)))  float  v8f;

union FragBF { v16bf v; uint4 q[2]; };

__device__ __forceinline__ uint16_t f2bf(float f) {
  uint32_t u = __float_as_uint(f);
  return (uint16_t)((u + 0x7FFFu + ((u >> 16) & 1u)) >> 16);
}
__device__ __forceinline__ float bf2f(uint16_t h) {
  return __uint_as_float(((uint32_t)h) << 16);
}
__device__ __forceinline__ float rowmax16(float v) {
#pragma unroll
  for (int m = 1; m < 16; m <<= 1) v = fmaxf(v, __shfl_xor(v, m, 32));
  return v;
}
__device__ __forceinline__ float rowsum16(float v) {
#pragma unroll
  for (int m = 1; m < 16; m <<= 1) v += __shfl_xor(v, m, 32);
  return v;
}

// -------- CDNA5 async global->LDS copy (ASYNCcnt-tracked) -------------------
__device__ __forceinline__ uint32_t lds_off(const void* p) {
  // LDS aperture flat address: addr[31:0] is the LDS byte offset.
  return (uint32_t)(size_t)p;
}
__device__ __forceinline__ void async_b128(uint32_t dst_lds, const void* src) {
  asm volatile("global_load_async_to_lds_b128 %0, %1, off"
               :
               : "v"(dst_lds), "v"((uint64_t)(size_t)src)
               : "memory");
}
__device__ __forceinline__ void wait_async0() {
  asm volatile("s_wait_asynccnt 0" ::: "memory");
}

__device__ __forceinline__ v8f wmma_bf16(const FragBF& a, const FragBF& b, v8f c) {
  return __builtin_amdgcn_wmma_f32_16x16x32_bf16(false, a.v, false, b.v,
                                                 (short)0, c, false, false);
}

// ---------------------------------------------------------------------------
// Elementwise / small kernels
// ---------------------------------------------------------------------------
__global__ void zero_counts_kernel(int* c) {
  if (threadIdx.x < 8) c[threadIdx.x] = 0;
}

__global__ void cast_bf16_kernel(const float* __restrict__ s,
                                 uint16_t* __restrict__ d, size_t n) {
  size_t i = (size_t)blockIdx.x * blockDim.x + threadIdx.x;
  size_t st = (size_t)gridDim.x * blockDim.x;
  for (; i < n; i += st) d[i] = f2bf(s[i]);
}

// one block per token: rmsnorm(prev,g1) -> out[:,0:1024], rmsnorm(fut,g2) -> out[:,1024:2048]
__global__ __launch_bounds__(256)
void rmsnorm_combine_kernel(const float* __restrict__ prev,
                            const float* __restrict__ fut,
                            const float* __restrict__ g1,
                            const float* __restrict__ g2,
                            uint16_t* __restrict__ out) {
  const int t = blockIdx.x, tid = threadIdx.x;
  __shared__ float red[256];
  const float* xs[2] = {prev, fut};
  const float* gs[2] = {g1, g2};
#pragma unroll
  for (int half = 0; half < 2; ++half) {
    const float* x = xs[half] + (size_t)t * 1024;
    float4 v = *(const float4*)&x[tid * 4];
    red[tid] = v.x * v.x + v.y * v.y + v.z * v.z + v.w * v.w;
    __syncthreads();
    for (int s = 128; s > 0; s >>= 1) {
      if (tid < s) red[tid] += red[tid + s];
      __syncthreads();
    }
    float inv = rsqrtf(red[0] * (1.0f / 1024.0f) + 1e-6f);
    __syncthreads();
    float4 g = *(const float4*)&gs[half][tid * 4];
    uint16_t* o = out + (size_t)t * 2048 + half * 1024 + tid * 4;
    o[0] = f2bf(v.x * inv * g.x);
    o[1] = f2bf(v.y * inv * g.y);
    o[2] = f2bf(v.z * inv * g.z);
    o[3] = f2bf(v.w * inv * g.w);
  }
}

// one block per token: out_bf16 = rmsnorm(x, g)
__global__ __launch_bounds__(256)
void rmsnorm_kernel(const float* __restrict__ x, const float* __restrict__ g,
                    uint16_t* __restrict__ out) {
  const int t = blockIdx.x, tid = threadIdx.x;
  __shared__ float red[256];
  const float* xr = x + (size_t)t * 1024;
  float4 v = *(const float4*)&xr[tid * 4];
  red[tid] = v.x * v.x + v.y * v.y + v.z * v.z + v.w * v.w;
  __syncthreads();
  for (int s = 128; s > 0; s >>= 1) {
    if (tid < s) red[tid] += red[tid + s];
    __syncthreads();
  }
  float inv = rsqrtf(red[0] * (1.0f / 1024.0f) + 1e-6f);
  float4 gv = *(const float4*)&g[tid * 4];
  uint16_t* o = out + (size_t)t * 1024 + tid * 4;
  o[0] = f2bf(v.x * inv * gv.x);
  o[1] = f2bf(v.y * inv * gv.y);
  o[2] = f2bf(v.z * inv * gv.z);
  o[3] = f2bf(v.w * inv * gv.w);
}

// ---------------------------------------------------------------------------
// Generic bf16 WMMA GEMM: C[M,N] = A[M,K] @ B[K,N]  (row-major, bf16 in)
// Block tile 128x64x32, 256 threads = 8 waves, each wave -> 32x32 (4 WMMA).
// A-tiles: double-buffered async global->LDS. B-tiles: VGPR transpose stage.
// ---------------------------------------------------------------------------
template <bool OUT_BF16, bool RESIDUAL>
__global__ __launch_bounds__(256)
void gemm_bf16_kernel(const uint16_t* __restrict__ A,
                      const uint16_t* __restrict__ B,
                      float* __restrict__ Cf, uint16_t* __restrict__ Cb,
                      int M, int N, int K) {
  __shared__ uint16_t As[2][128 * 40];   // [row][k], pad 40
  __shared__ uint16_t Bs[2][64 * 40];    // transposed: [n][k], pad 40
  const int tid = threadIdx.x;
  const int wave = tid >> 5, lane = tid & 31;
  const int bm = blockIdx.x * 128, bn = blockIdx.y * 64;
  const int wm = (wave & 3) * 32;        // wave row offset within tile
  const int wn = (wave >> 2) * 32;       // wave col offset within tile
  v8f c00 = {}, c01 = {}, c10 = {}, c11 = {};

  const int arow = tid >> 1;             // 0..127
  const int acol = (tid & 1) * 16;       // 0 or 16
  const int bkr = tid >> 3;              // 0..31
  const int bcol = (tid & 7) * 8;        // 0..56
  const size_t abase = (size_t)(bm + arow) * K + acol;

  // prologue: tile 0 in flight
  async_b128(lds_off(&As[0][arow * 40 + acol]), &A[abase]);
  async_b128(lds_off(&As[0][arow * 40 + acol + 8]), &A[abase + 8]);
  uint4 breg = *(const uint4*)&B[(size_t)bkr * N + bn + bcol];

  const int ntile = K >> 5;
  for (int i = 0; i < ntile; ++i) {
    const int buf = i & 1;
    // stage B tile i into LDS (transposed to n-major)
    {
      const uint16_t* bh = (const uint16_t*)&breg;
#pragma unroll
      for (int j = 0; j < 8; ++j) Bs[buf][(bcol + j) * 40 + bkr] = bh[j];
    }
    wait_async0();
    __syncthreads();
    // kick off tile i+1 while computing tile i
    if (i + 1 < ntile) {
      const int k0 = (i + 1) * 32;
      async_b128(lds_off(&As[buf ^ 1][arow * 40 + acol]), &A[abase + k0]);
      async_b128(lds_off(&As[buf ^ 1][arow * 40 + acol + 8]), &A[abase + k0 + 8]);
      breg = *(const uint4*)&B[(size_t)(k0 + bkr) * N + bn + bcol];
    }
    FragBF a0, a1, b0, b1;
    const int m = lane & 15;
    const int kh = (lane >> 4) * 16;
    const int kb = (lane >> 4) * 16;
    a0.q[0] = *(const uint4*)&As[buf][(wm + m) * 40 + kh];
    a0.q[1] = *(const uint4*)&As[buf][(wm + m) * 40 + kh + 8];
    a1.q[0] = *(const uint4*)&As[buf][(wm + 16 + m) * 40 + kh];
    a1.q[1] = *(const uint4*)&As[buf][(wm + 16 + m) * 40 + kh + 8];
    b0.q[0] = *(const uint4*)&Bs[buf][(wn + m) * 40 + kb];
    b0.q[1] = *(const uint4*)&Bs[buf][(wn + m) * 40 + kb + 8];
    b1.q[0] = *(const uint4*)&Bs[buf][(wn + 16 + m) * 40 + kb];
    b1.q[1] = *(const uint4*)&Bs[buf][(wn + 16 + m) * 40 + kb + 8];
    c00 = wmma_bf16(a0, b0, c00);
    c01 = wmma_bf16(a0, b1, c01);
    c10 = wmma_bf16(a1, b0, c10);
    c11 = wmma_bf16(a1, b1, c11);
  }

  const int mrow = bm + wm + (lane >> 4) * 8;
  const int ncol = bn + wn + (lane & 15);
#pragma unroll
  for (int r = 0; r < 8; ++r) {
    size_t i00 = (size_t)(mrow + r) * N + ncol;          // rows wm..wm+15
    size_t i10 = (size_t)(mrow + 16 + r) * N + ncol;     // rows wm+16..wm+31
    float v00 = c00[r], v01 = c01[r], v10 = c10[r], v11 = c11[r];
    if (RESIDUAL) {
      v00 += Cf[i00]; v01 += Cf[i00 + 16];
      v10 += Cf[i10]; v11 += Cf[i10 + 16];
    }
    if (OUT_BF16) {
      Cb[i00] = f2bf(v00); Cb[i00 + 16] = f2bf(v01);
      Cb[i10] = f2bf(v10); Cb[i10 + 16] = f2bf(v11);
    } else {
      Cf[i00] = v00; Cf[i00 + 16] = v01;
      Cf[i10] = v10; Cf[i10 + 16] = v11;
    }
  }
}

// ---------------------------------------------------------------------------
// Flash MLA attention: per (qtile=128, head, batch). 8 waves, 16 q-rows/wave.
// q,k,v: bf16 [B*S, H*64]; out: bf16 [B*S, H*64]
// ---------------------------------------------------------------------------
__global__ __launch_bounds__(256)
void mla_flash_kernel(const uint16_t* __restrict__ Q,
                      const uint16_t* __restrict__ Kg,
                      const uint16_t* __restrict__ Vg,
                      uint16_t* __restrict__ O, int S) {
  __shared__ uint16_t Ks[64 * 72];        // [key][hd]  (B n-major for QK^T)
  __shared__ uint16_t Vs[64 * 72];        // [hd][key]  (B n-major for PV)
  __shared__ uint16_t Ps[8][16 * 72];     // per-wave P tile [qrow][key]

  const int tid = threadIdx.x;
  const int wave = tid >> 5, lane = tid & 31;
  const int h = blockIdx.y, bz = blockIdx.z;
  const int qlo = blockIdx.x * 128;
  const int qbase = qlo + wave * 16;

  FragBF qa[2];
  {
    const int qrow = qbase + (lane & 15);
    size_t qoff = ((size_t)bz * S + qrow) * 1024 + h * 64;
#pragma unroll
    for (int ks = 0; ks < 2; ++ks) {
      int kh = (lane >> 4) * 16 + ks * 32;
      qa[ks].q[0] = *(const uint4*)&Q[qoff + kh];
      qa[ks].q[1] = *(const uint4*)&Q[qoff + kh + 8];
    }
  }

  v8f of[4] = {};
  float mrun[8], lrun[8];
#pragma unroll
  for (int r = 0; r < 8; ++r) { mrun[r] = -1e30f; lrun[r] = 0.0f; }

  const int nkt = qlo / 64 + 2;           // causal tile bound (exclusive)
  for (int kt = 0; kt < nkt; ++kt) {
    // K tile via async copy; V tile via VGPR transpose
    {
      const int row = tid >> 2;           // key 0..63
      const int col = (tid & 3) * 16;     // hd base
      size_t gb = ((size_t)bz * S + kt * 64 + row) * 1024 + h * 64 + col;
      async_b128(lds_off(&Ks[row * 72 + col]), &Kg[gb]);
      async_b128(lds_off(&Ks[row * 72 + col + 8]), &Kg[gb + 8]);
      uint4 v0 = *(const uint4*)&Vg[gb];
      uint4 v1 = *(const uint4*)&Vg[gb + 8];
      const uint16_t* vh0 = (const uint16_t*)&v0;
      const uint16_t* vh1 = (const uint16_t*)&v1;
#pragma unroll
      for (int i = 0; i < 8; ++i) {
        Vs[(col + i) * 72 + row] = vh0[i];
        Vs[(col + 8 + i) * 72 + row] = vh1[i];
      }
    }
    wait_async0();
    __syncthreads();

    if (kt * 64 <= qbase + 15) {
      // ---- scores: S16x64 = Q(16x64) x K^T ----
      v8f sf[4];
#pragma unroll
      for (int n = 0; n < 4; ++n) {
        sf[n] = v8f{};
#pragma unroll
        for (int ks = 0; ks < 2; ++ks) {
          FragBF bk;
          int base = (n * 16 + (lane & 15)) * 72 + ks * 32 + (lane >> 4) * 16;
          bk.q[0] = *(const uint4*)&Ks[base];
          bk.q[1] = *(const uint4*)&Ks[base + 8];
          sf[n] = wmma_bf16(qa[ks], bk, sf[n]);
        }
      }
      // scale + causal mask
#pragma unroll
      for (int n = 0; n < 4; ++n) {
        int key = kt * 64 + n * 16 + (lane & 15);
#pragma unroll
        for (int r = 0; r < 8; ++r) {
          int query = qbase + r + 8 * (lane >> 4);
          float s = sf[n][r] * 0.125f;
          sf[n][r] = (key > query) ? -1e30f : s;
        }
      }
      // online softmax (row reductions via shfl within 16-lane halves)
#pragma unroll
      for (int r = 0; r < 8; ++r) {
        float tm = fmaxf(fmaxf(sf[0][r], sf[1][r]), fmaxf(sf[2][r], sf[3][r]));
        tm = rowmax16(tm);
        float mn = fmaxf(mrun[r], tm);
        float alpha = __expf(mrun[r] - mn);
        float psum = 0.0f;
#pragma unroll
        for (int n = 0; n < 4; ++n) {
          float p = __expf(sf[n][r] - mn);
          sf[n][r] = p;
          psum += p;
        }
        psum = rowsum16(psum);
        lrun[r] = lrun[r] * alpha + psum;
        mrun[r] = mn;
#pragma unroll
        for (int n = 0; n < 4; ++n) of[n][r] *= alpha;
      }
      // stage P (C-layout -> A-layout via per-wave LDS)
#pragma unroll
      for (int n = 0; n < 4; ++n)
#pragma unroll
        for (int r = 0; r < 8; ++r) {
          int row = r + 8 * (lane >> 4);
          Ps[wave][row * 72 + n * 16 + (lane & 15)] = f2bf(sf[n][r]);
        }
      // ---- O += P(16x64) x V(64x64) ----
#pragma unroll
      for (int ks = 0; ks < 2; ++ks) {
        FragBF pa;
        int m = lane & 15;
        int kh = (lane >> 4) * 16 + ks * 32;
        pa.q[0] = *(const uint4*)&Ps[wave][m * 72 + kh];
        pa.q[1] = *(const uint4*)&Ps[wave][m * 72 + kh + 8];
#pragma unroll
        for (int n = 0; n < 4; ++n) {
          FragBF bv;
          int base = (n * 16 + (lane & 15)) * 72 + ks * 32 + (lane >> 4) * 16;
          bv.q[0] = *(const uint4*)&Vs[base];
          bv.q[1] = *(const uint4*)&Vs[base + 8];
          of[n] = wmma_bf16(pa, bv, of[n]);
        }
      }
    }
    __syncthreads();
  }

  // normalize + store
#pragma unroll
  for (int r = 0; r < 8; ++r) {
    float inv = 1.0f / lrun[r];
    int token = qbase + r + 8 * (lane >> 4);
    size_t base = ((size_t)bz * S + token) * 1024 + h * 64 + (lane & 15);
#pragma unroll
    for (int n = 0; n < 4; ++n) O[base + n * 16] = f2bf(of[n][r] * inv);
  }
}

// ---------------------------------------------------------------------------
// MoE gating: one block per token; softmax over 8 logits, top-2, atomic append.
// ---------------------------------------------------------------------------
__global__ __launch_bounds__(256)
void moe_gate_kernel(const uint16_t* __restrict__ X,
                     const float* __restrict__ WG, int* __restrict__ counts,
                     int* __restrict__ idxb, float* __restrict__ gateb, int T) {
  const int t = blockIdx.x, tid = threadIdx.x;
  float acc[8] = {};
  for (int i = tid; i < 1024; i += 256) {
    float xv = bf2f(X[(size_t)t * 1024 + i]);
#pragma unroll
    for (int e = 0; e < 8; ++e) acc[e] += xv * WG[i * 8 + e];
  }
  __shared__ float red[256];
  __shared__ float logits[8];
#pragma unroll
  for (int e = 0; e < 8; ++e) {
    red[tid] = acc[e];
    __syncthreads();
    for (int s = 128; s > 0; s >>= 1) {
      if (tid < s) red[tid] += red[tid + s];
      __syncthreads();
    }
    if (tid == 0) logits[e] = red[0];
    __syncthreads();
  }
  if (tid == 0) {
    float mx = -1e30f;
#pragma unroll
    for (int e = 0; e < 8; ++e) mx = fmaxf(mx, logits[e]);
    float p[8], se = 0.0f;
#pragma unroll
    for (int e = 0; e < 8; ++e) { p[e] = __expf(logits[e] - mx); se += p[e]; }
    float inv = 1.0f / se;
#pragma unroll
    for (int e = 0; e < 8; ++e) p[e] *= inv;
    int i0 = 0;
#pragma unroll
    for (int e = 1; e < 8; ++e) if (p[e] > p[i0]) i0 = e;
    int i1 = (i0 == 0) ? 1 : 0;
#pragma unroll
    for (int e = 0; e < 8; ++e) if (e != i0 && p[e] > p[i1]) i1 = e;
    float s2 = 1.0f / (p[i0] + p[i1]);
    int s0 = atomicAdd(&counts[i0], 1);
    idxb[i0 * T + s0] = t;
    gateb[i0 * T + s0] = p[i0] * s2;
    int s1 = atomicAdd(&counts[i1], 1);
    idxb[i1 * T + s1] = t;
    gateb[i1 * T + s1] = p[i1] * s2;
  }
}

// ---------------------------------------------------------------------------
// MoE up: H[slot, F] = silu(Xg @ W1) * (Xg @ W3), Xg gathered by idx list.
// ---------------------------------------------------------------------------
__global__ __launch_bounds__(256)
void moe_up_kernel(const uint16_t* __restrict__ X,   // [T, D]
                   const uint16_t* __restrict__ W1,  // [D, F]
                   const uint16_t* __restrict__ W3,  // [D, F]
                   const int* __restrict__ idx, const int* __restrict__ count,
                   uint16_t* __restrict__ H,         // [T, F] by slot
                   int D, int F) {
  const int cnt = *count;
  if (blockIdx.x * 64 >= cnt) return;
  __shared__ uint16_t As[64 * 40];
  __shared__ uint16_t B1s[64 * 40];
  __shared__ uint16_t B3s[64 * 40];
  const int tid = threadIdx.x;
  const int wave = tid >> 5, lane = tid & 31;
  const int bm = blockIdx.x * 64, bn = blockIdx.y * 64;
  const int wm = (wave & 3) * 16, wn = (wave >> 2) * 32;
  v8f c10 = {}, c11 = {}, c30 = {}, c31 = {};

  const int arow = tid >> 2, acol = (tid & 3) * 8;
  const int bkr = tid >> 3, bcol = (tid & 7) * 8;
  const int aslot = bm + arow;
  const int agrow = (aslot < cnt) ? idx[aslot] : 0;

  for (int k0 = 0; k0 < D; k0 += 32) {
    async_b128(lds_off(&As[arow * 40 + acol]),
               &X[(size_t)agrow * D + k0 + acol]);
    uint4 b1 = *(const uint4*)&W1[(size_t)(k0 + bkr) * F + bn + bcol];
    uint4 b3 = *(const uint4*)&W3[(size_t)(k0 + bkr) * F + bn + bcol];
    const uint16_t* h1 = (const uint16_t*)&b1;
    const uint16_t* h3 = (const uint16_t*)&b3;
#pragma unroll
    for (int i = 0; i < 8; ++i) {
      B1s[(bcol + i) * 40 + bkr] = h1[i];
      B3s[(bcol + i) * 40 + bkr] = h3[i];
    }
    wait_async0();
    __syncthreads();

    FragBF a, b10, b11, b30, b31;
    const int m = lane & 15;
    const int kh = (lane >> 4) * 16;
    const int kb = (lane >> 4) * 16;
    a.q[0] = *(const uint4*)&As[(wm + m) * 40 + kh];
    a.q[1] = *(const uint4*)&As[(wm + m) * 40 + kh + 8];
    b10.q[0] = *(const uint4*)&B1s[(wn + m) * 40 + kb];
    b10.q[1] = *(const uint4*)&B1s[(wn + m) * 40 + kb + 8];
    b11.q[0] = *(const uint4*)&B1s[(wn + 16 + m) * 40 + kb];
    b11.q[1] = *(const uint4*)&B1s[(wn + 16 + m) * 40 + kb + 8];
    b30.q[0] = *(const uint4*)&B3s[(wn + m) * 40 + kb];
    b30.q[1] = *(const uint4*)&B3s[(wn + m) * 40 + kb + 8];
    b31.q[0] = *(const uint4*)&B3s[(wn + 16 + m) * 40 + kb];
    b31.q[1] = *(const uint4*)&B3s[(wn + 16 + m) * 40 + kb + 8];
    c10 = wmma_bf16(a, b10, c10);
    c11 = wmma_bf16(a, b11, c11);
    c30 = wmma_bf16(a, b30, c30);
    c31 = wmma_bf16(a, b31, c31);
    __syncthreads();
  }

  const int srow = bm + wm + (lane >> 4) * 8;
  const int ncol = bn + wn + (lane & 15);
#pragma unroll
  for (int r = 0; r < 8; ++r) {
    int slot = srow + r;
    if (slot < cnt) {
      float g0 = c10[r], g1 = c11[r];
      float h0 = (g0 / (1.0f + __expf(-g0))) * c30[r];
      float h1 = (g1 / (1.0f + __expf(-g1))) * c31[r];
      H[(size_t)slot * F + ncol] = f2bf(h0);
      H[(size_t)slot * F + ncol + 16] = f2bf(h1);
    }
  }
}

// ---------------------------------------------------------------------------
// MoE down: OUT[idx[slot], :] += gate[slot] * (H[slot, :] @ W2)
// ---------------------------------------------------------------------------
__global__ __launch_bounds__(256)
void moe_down_kernel(const uint16_t* __restrict__ H,  // [T, F] by slot
                     const uint16_t* __restrict__ W2, // [F, D]
                     const int* __restrict__ idx, const float* __restrict__ gate,
                     const int* __restrict__ count, float* __restrict__ OUT,
                     int F, int D) {
  const int cnt = *count;
  if (blockIdx.x * 64 >= cnt) return;
  __shared__ uint16_t As[64 * 40];
  __shared__ uint16_t Bs[64 * 40];
  const int tid = threadIdx.x;
  const int wave = tid >> 5, lane = tid & 31;
  const int bm = blockIdx.x * 64, bn = blockIdx.y * 64;
  const int wm = (wave & 3) * 16, wn = (wave >> 2) * 32;
  v8f c0 = {}, c1 = {};

  const int arow = tid >> 2, acol = (tid & 3) * 8;
  const int bkr = tid >> 3, bcol = (tid & 7) * 8;

  for (int k0 = 0; k0 < F; k0 += 32) {
    async_b128(lds_off(&As[arow * 40 + acol]),
               &H[(size_t)(bm + arow) * F + k0 + acol]);
    uint4 bv = *(const uint4*)&W2[(size_t)(k0 + bkr) * D + bn + bcol];
    const uint16_t* bh = (const uint16_t*)&bv;
#pragma unroll
    for (int i = 0; i < 8; ++i) Bs[(bcol + i) * 40 + bkr] = bh[i];
    wait_async0();
    __syncthreads();

    FragBF a, b0, b1;
    const int m = lane & 15;
    const int kh = (lane >> 4) * 16;
    const int kb = (lane >> 4) * 16;
    a.q[0] = *(const uint4*)&As[(wm + m) * 40 + kh];
    a.q[1] = *(const uint4*)&As[(wm + m) * 40 + kh + 8];
    b0.q[0] = *(const uint4*)&Bs[(wn + m) * 40 + kb];
    b0.q[1] = *(const uint4*)&Bs[(wn + m) * 40 + kb + 8];
    b1.q[0] = *(const uint4*)&Bs[(wn + 16 + m) * 40 + kb];
    b1.q[1] = *(const uint4*)&Bs[(wn + 16 + m) * 40 + kb + 8];
    c0 = wmma_bf16(a, b0, c0);
    c1 = wmma_bf16(a, b1, c1);
    __syncthreads();
  }

  const int srow = bm + wm + (lane >> 4) * 8;
  const int ncol = bn + wn + (lane & 15);
#pragma unroll
  for (int r = 0; r < 8; ++r) {
    int slot = srow + r;
    if (slot < cnt) {
      int token = idx[slot];
      float sc = gate[slot];
      size_t i0 = (size_t)token * D + ncol;
      OUT[i0] += sc * c0[r];
      OUT[i0 + 16] += sc * c1[r];
    }
  }
}

// ---------------------------------------------------------------------------
// Host launcher
// ---------------------------------------------------------------------------
extern "C" void kernel_launch(void* const* d_in, const int* in_sizes, int n_in,
                              void* d_out, int out_size, void* d_ws,
                              size_t ws_size, hipStream_t stream) {
  const float* prev   = (const float*)d_in[0];
  const float* fut    = (const float*)d_in[1];
  const float* w_comb = (const float*)d_in[2];
  const float* g1     = (const float*)d_in[3];
  const float* g2     = (const float*)d_in[4];
  const float* g_attn = (const float*)d_in[5];
  const float* g_mlp  = (const float*)d_in[6];
  const float* wq     = (const float*)d_in[7];
  const float* w_dkv  = (const float*)d_in[8];
  const float* w_uk   = (const float*)d_in[9];
  const float* w_uv   = (const float*)d_in[10];
  const float* wo     = (const float*)d_in[11];
  const float* w_gate = (const float*)d_in[12];
  const float* w1     = (const float*)d_in[13];
  const float* w3     = (const float*)d_in[14];
  const float* w2     = (const float*)d_in[15];

  constexpr int Bb = 4, S = 2048, D = 1024, L = 512, E = 8, F = 2048, Hh = 16;
  constexpr int T = Bb * S;

  uint8_t* ws = (uint8_t*)d_ws;
  size_t off = 0;
  auto alloc = [&](size_t bytes) -> uint8_t* {
    uint8_t* p = ws + off;
    off = (off + bytes + 255) & ~(size_t)255;
    return p;
  };
  uint16_t* wcb   = (uint16_t*)alloc((size_t)2 * D * D * 2);
  uint16_t* wqb   = (uint16_t*)alloc((size_t)D * D * 2);
  uint16_t* wdkvb = (uint16_t*)alloc((size_t)D * L * 2);
  uint16_t* wukb  = (uint16_t*)alloc((size_t)L * D * 2);
  uint16_t* wuvb  = (uint16_t*)alloc((size_t)L * D * 2);
  uint16_t* wob   = (uint16_t*)alloc((size_t)D * D * 2);
  uint16_t* w1b   = (uint16_t*)alloc((size_t)E * D * F * 2);
  uint16_t* w3b   = (uint16_t*)alloc((size_t)E * D * F * 2);
  uint16_t* w2b   = (uint16_t*)alloc((size_t)E * F * D * 2);
  uint16_t* comb  = (uint16_t*)alloc((size_t)T * 2 * D * 2);
  float*    hid   = (float*)   alloc((size_t)T * D * 4);
  uint16_t* xn    = (uint16_t*)alloc((size_t)T * D * 2);
  uint16_t* qb    = (uint16_t*)alloc((size_t)T * D * 2);
  uint16_t* cb    = (uint16_t*)alloc((size_t)T * L * 2);
  uint16_t* kb    = (uint16_t*)alloc((size_t)T * D * 2);
  uint16_t* vb    = (uint16_t*)alloc((size_t)T * D * 2);
  uint16_t* ao    = (uint16_t*)alloc((size_t)T * D * 2);
  uint16_t* hbuf  = (uint16_t*)alloc((size_t)T * F * 2);
  int*      counts= (int*)     alloc((size_t)E * 4);
  int*      idxb  = (int*)     alloc((size_t)E * T * 4);
  float*    gateb = (float*)   alloc((size_t)E * T * 4);
  float*    outp  = (float*)d_out;

  zero_counts_kernel<<<1, 32, 0, stream>>>(counts);

  // weights -> bf16
  cast_bf16_kernel<<<2048, 256, 0, stream>>>(w_comb, wcb, (size_t)2 * D * D);
  cast_bf16_kernel<<<2048, 256, 0, stream>>>(wq, wqb, (size_t)D * D);
  cast_bf16_kernel<<<2048, 256, 0, stream>>>(w_dkv, wdkvb, (size_t)D * L);
  cast_bf16_kernel<<<2048, 256, 0, stream>>>(w_uk, wukb, (size_t)L * D);
  cast_bf16_kernel<<<2048, 256, 0, stream>>>(w_uv, wuvb, (size_t)L * D);
  cast_bf16_kernel<<<2048, 256, 0, stream>>>(wo, wob, (size_t)D * D);
  cast_bf16_kernel<<<4096, 256, 0, stream>>>(w1, w1b, (size_t)E * D * F);
  cast_bf16_kernel<<<4096, 256, 0, stream>>>(w3, w3b, (size_t)E * D * F);
  cast_bf16_kernel<<<4096, 256, 0, stream>>>(w2, w2b, (size_t)E * F * D);

  // 1) dual rmsnorm + concat (bf16)
  rmsnorm_combine_kernel<<<T, 256, 0, stream>>>(prev, fut, g1, g2, comb);
  // 2) hidden = combined @ w_combine   [T,2048]x[2048,1024]
  gemm_bf16_kernel<false, false><<<dim3(T / 128, D / 64), 256, 0, stream>>>(
      comb, wcb, hid, nullptr, T, D, 2 * D);
  // 3) attention input norm
  rmsnorm_kernel<<<T, 256, 0, stream>>>(hid, g_attn, xn);
  // 4) projections
  gemm_bf16_kernel<true, false><<<dim3(T / 128, D / 64), 256, 0, stream>>>(
      xn, wqb, nullptr, qb, T, D, D);
  gemm_bf16_kernel<true, false><<<dim3(T / 128, L / 64), 256, 0, stream>>>(
      xn, wdkvb, nullptr, cb, T, L, D);
  gemm_bf16_kernel<true, false><<<dim3(T / 128, D / 64), 256, 0, stream>>>(
      cb, wukb, nullptr, kb, T, D, L);
  gemm_bf16_kernel<true, false><<<dim3(T / 128, D / 64), 256, 0, stream>>>(
      cb, wuvb, nullptr, vb, T, D, L);
  // 5) flash MLA attention
  mla_flash_kernel<<<dim3(S / 128, Hh, Bb), 256, 0, stream>>>(qb, kb, vb, ao, S);
  // 6) hidden += attn @ wo
  gemm_bf16_kernel<false, true><<<dim3(T / 128, D / 64), 256, 0, stream>>>(
      ao, wob, hid, nullptr, T, D, D);
  // 7) MoE input norm + gating
  rmsnorm_kernel<<<T, 256, 0, stream>>>(hid, g_mlp, xn);
  hipMemcpyAsync(outp, hid, (size_t)T * D * sizeof(float),
                 hipMemcpyDeviceToDevice, stream);
  moe_gate_kernel<<<T, 256, 0, stream>>>(xn, w_gate, counts, idxb, gateb, T);
  // 8) experts (top-2 sparse)
  for (int e = 0; e < E; ++e) {
    moe_up_kernel<<<dim3(T / 64, F / 64), 256, 0, stream>>>(
        xn, w1b + (size_t)e * D * F, w3b + (size_t)e * D * F, idxb + e * T,
        counts + e, hbuf, D, F);
    moe_down_kernel<<<dim3(T / 64, D / 64), 256, 0, stream>>>(
        hbuf, w2b + (size_t)e * F * D, idxb + e * T, gateb + e * T, counts + e,
        outp, F, D);
  }
}
